// Encoder_65635690218112
// MI455X (gfx1250) — matrix operand order repaired and checked by reading, not compiled, not run on hardware
//
#include <hip/hip_runtime.h>

#define DEV __device__ __forceinline__

typedef __attribute__((ext_vector_type(8)))  float  v8f;
typedef __attribute__((ext_vector_type(16))) __bf16 v16bf;
typedef __attribute__((ext_vector_type(8)))  __bf16 v8bf;

// ---- problem sizes ----
#define BB    16
#define CIN   80
#define TT    4096
#define DD    1024
#define LL    6
#define COUT  194
#define KPRE  96            // CIN padded up to multiple of 32
#define NPOST 208           // COUT padded up to multiple of 16
#define CL    64            // scan chunk length
#define NCHK  (TT / CL)     // 64 chunks
#define LDA   40            // LDS row stride in bf16 (80B) -> conflict-free b128 frag reads

// ============================================================
// WMMA fragment loaders (ISA 7.12.2 layouts). LDS tiles row
// stride = LDA bf16.
// ============================================================
// A matrix 16x32 bf16: lanes 0-15 row M=lane, K = 0..7 & 16..23
//                      lanes 16-31 row M=lane-16, K = 8..15 & 24..31
DEV v16bf load_fragA(const __bf16* base, int row0, int lane) {
  const int r  = row0 + (lane & 15);
  const int k0 = (lane & 16) ? 8 : 0;
  const __bf16* p = base + r * LDA + k0;
  v8bf lo = *(const v8bf*)(p);
  v8bf hi = *(const v8bf*)(p + 16);
  v16bf o;
#pragma unroll
  for (int i = 0; i < 8; ++i) { o[i] = lo[i]; o[i + 8] = hi[i]; }
  return o;
}
// B matrix 32x16 bf16 (LDS holds it transposed, [n][k]):
// lanes 0-15 col N=lane, K = 0..7 & 16..23; lanes 16-31 col N=lane-16, K = 8..15 & 24..31
DEV v16bf load_fragB(const __bf16* base, int col0, int lane) {
  const int n  = col0 + (lane & 15);
  const int k0 = (lane & 16) ? 8 : 0;
  const __bf16* p = base + n * LDA + k0;
  v8bf lo = *(const v8bf*)(p);
  v8bf hi = *(const v8bf*)(p + 16);
  v16bf o;
#pragma unroll
  for (int i = 0; i < 8; ++i) { o[i] = lo[i]; o[i + 8] = hi[i]; }
  return o;
}

DEV v8f wmma_bf16(v16bf a, v16bf b, v8f c) {
  return __builtin_amdgcn_wmma_f32_16x16x32_bf16(
      /*neg_a=*/false, a, /*neg_b=*/false, b,
      /*c_mod=*/(short)0, c, /*reuse_a=*/false, /*reuse_b=*/false);
}

DEV float sigm(float x) { return 1.0f / (1.0f + __expf(-x)); }

// pack 16 floats -> two v8bf
DEV void pack16(const float* v, v8bf& p0, v8bf& p1) {
#pragma unroll
  for (int i = 0; i < 8; ++i) { p0[i] = (__bf16)v[i]; p1[i] = (__bf16)v[i + 8]; }
}

// ============================================================
// Weight prep: fp32 -> bf16, transposed to [n][k]
// ============================================================
__global__ void prep_layer_w(const float* __restrict__ W, __bf16* __restrict__ Wt) {
  // W: [LL][DD][2*DD]  ->  Wt: [LL][2*DD][DD]
  size_t idx = (size_t)blockIdx.x * 256 + threadIdx.x;
  const size_t per = (size_t)2 * DD * DD;
  int l = (int)(idx / per);
  size_t r = idx % per;
  int n = (int)(r / DD), k = (int)(r % DD);
  Wt[idx] = (__bf16)W[((size_t)l * DD + k) * (2 * DD) + n];
}

__global__ void prep_pre_w(const float* __restrict__ W, __bf16* __restrict__ Wt) {
  // W: [CIN][DD] -> Wt: [DD][KPRE] (zero-padded K)
  size_t idx = (size_t)blockIdx.x * 256 + threadIdx.x;
  int n = (int)(idx / KPRE), k = (int)(idx % KPRE);
  Wt[idx] = (k < CIN) ? (__bf16)W[(size_t)k * DD + n] : (__bf16)0.0f;
}

__global__ void prep_post_w(const float* __restrict__ W, __bf16* __restrict__ Wt) {
  // W: [DD][COUT] -> Wt: [NPOST][DD] (zero-padded N)
  size_t idx = (size_t)blockIdx.x * 256 + threadIdx.x;
  int n = (int)(idx / DD), k = (int)(idx % DD);
  Wt[idx] = (n < COUT) ? (__bf16)W[(size_t)k * COUT + n] : (__bf16)0.0f;
}

// ============================================================
// Pre-GEMM: h[b][t][d] = sum_c x[b][c][t] * w_pre[c][d] + b_pre[d]
// M = B*T (tile 128), N = DD (tile 128), K = KPRE.
// ============================================================
__global__ __launch_bounds__(256)
void gemm_pre(const float* __restrict__ X, const __bf16* __restrict__ Wt,
              const float* __restrict__ bias, float* __restrict__ H) {
  const int tid = threadIdx.x, lane = tid & 31, wv = tid >> 5;
  const int mIdx = wv >> 1, nIdx = wv & 1;
  const size_t m0 = (size_t)blockIdx.x * 128;
  const int n0 = blockIdx.y * 128;
  const int b = (int)(m0 / TT), t0 = (int)(m0 % TT);

  __shared__ __align__(16) __bf16 lA[128][LDA];
  __shared__ __align__(16) __bf16 lB[128][LDA];

  v8f acc[2][4] = {};
  for (int kt = 0; kt < KPRE / 32; ++kt) {
    const int k0 = kt * 32;
    __syncthreads();
    { // A tile: transpose load from x[b][c][t]; batch all 16 loads, then pack+store
      const int tt = tid & 127, cg = tid >> 7;
      float v[16];
#pragma unroll
      for (int i = 0; i < 16; ++i) {
        const int c = k0 + cg * 16 + i;
        const int cc = (c < CIN) ? c : (CIN - 1);              // in-bounds clamp
        v[i] = X[((size_t)b * CIN + cc) * TT + t0 + tt];
      }
#pragma unroll
      for (int i = 0; i < 16; ++i)
        if (k0 + cg * 16 + i >= CIN) v[i] = 0.0f;              // branchless pad
      v8bf p0, p1; pack16(v, p0, p1);
      *(v8bf*)&lA[tt][cg * 16]     = p0;
      *(v8bf*)&lA[tt][cg * 16 + 8] = p1;
    }
    { // B tile: rows n0..n0+127 of Wt[DD][KPRE]
      const int rr = tid >> 1, half = tid & 1;
      const __bf16* src = Wt + (size_t)(n0 + rr) * KPRE + k0 + half * 16;
      uint4 u0 = *(const uint4*)(src);
      uint4 u1 = *(const uint4*)(src + 8);
      *(uint4*)&lB[rr][half * 16]     = u0;
      *(uint4*)&lB[rr][half * 16 + 8] = u1;
    }
    __syncthreads();
    v16bf af[2], bf[4];
#pragma unroll
    for (int mf = 0; mf < 2; ++mf) af[mf] = load_fragA(&lA[0][0], mIdx * 32 + mf * 16, lane);
#pragma unroll
    for (int nf = 0; nf < 4; ++nf) bf[nf] = load_fragB(&lB[0][0], nIdx * 64 + nf * 16, lane);
#pragma unroll
    for (int mf = 0; mf < 2; ++mf)
#pragma unroll
      for (int nf = 0; nf < 4; ++nf) acc[mf][nf] = wmma_bf16(af[mf], bf[nf], acc[mf][nf]);
  }
#pragma unroll
  for (int mf = 0; mf < 2; ++mf)
#pragma unroll
    for (int nf = 0; nf < 4; ++nf) {
      const int col = n0 + nIdx * 64 + nf * 16 + (lane & 15);
      const float bs = bias[col];
#pragma unroll
      for (int r = 0; r < 8; ++r) {
        const size_t row = m0 + mIdx * 32 + mf * 16 + ((lane < 16) ? r : r + 8);
        H[row * DD + col] = acc[mf][nf][r] + bs;
      }
    }
}

// ============================================================
// Layer GEMM + fused gate, double-buffered LDS.
//   zh = h @ W + b ; a = 1 - sigmoid(zh_z) ; bb = sigmoid(zh_z) * zh_h
// Each block covers 64 z-columns AND their matching 64 h~-columns.
// ============================================================
__global__ __launch_bounds__(256)
void gemm_layer(const float* __restrict__ H, const __bf16* __restrict__ Wt,
                const float* __restrict__ bias,
                float* __restrict__ Aout, float* __restrict__ Bout) {
  const int tid = threadIdx.x, lane = tid & 31, wv = tid >> 5;
  const int mIdx = wv >> 1, nIdx = wv & 1;
  const size_t m0 = (size_t)blockIdx.x * 128;
  const int n0 = blockIdx.y * 64;          // z-column base (h~ base = DD + n0)

  __shared__ __align__(16) __bf16 lA[2][128][LDA];
  __shared__ __align__(16) __bf16 lB[2][128][LDA];   // rows 0..63 z, 64..127 h~

  const int r = tid >> 1, half = tid & 1;
  const int gn = (r < 64) ? (n0 + r) : (DD + n0 + (r - 64));
  const float*  aSrc = H + (m0 + r) * DD + half * 16;
  const __bf16* bSrc = Wt + (size_t)gn * DD + half * 16;

  // staged fetch registers
  float4 fa[4];
  uint4  ub[2];
  // prologue: fetch k-slice 0
#pragma unroll
  for (int q = 0; q < 4; ++q) fa[q] = *(const float4*)(aSrc + 4 * q);
  ub[0] = *(const uint4*)(bSrc);
  ub[1] = *(const uint4*)(bSrc + 8);

  v8f acc[2][4] = {};
  int buf = 0;
  for (int kt = 0; kt < DD / 32; ++kt) {
    // stage regs -> LDS[buf]
    {
      float v[16];
#pragma unroll
      for (int q = 0; q < 4; ++q) {
        v[4 * q]     = fa[q].x; v[4 * q + 1] = fa[q].y;
        v[4 * q + 2] = fa[q].z; v[4 * q + 3] = fa[q].w;
      }
      v8bf p0, p1; pack16(v, p0, p1);
      *(v8bf*)&lA[buf][r][half * 16]     = p0;
      *(v8bf*)&lA[buf][r][half * 16 + 8] = p1;
      *(uint4*)&lB[buf][r][half * 16]     = ub[0];
      *(uint4*)&lB[buf][r][half * 16 + 8] = ub[1];
    }
    __syncthreads();
    // fetch next k-slice into regs (overlaps with WMMA below)
    if (kt + 1 < DD / 32) {
      const int k1 = (kt + 1) * 32;
#pragma unroll
      for (int q = 0; q < 4; ++q) fa[q] = *(const float4*)(aSrc + k1 + 4 * q);
      ub[0] = *(const uint4*)(bSrc + k1);
      ub[1] = *(const uint4*)(bSrc + k1 + 8);
      __builtin_prefetch(aSrc + k1 + 32, 0, 1);
    }
    // compute from LDS[buf]
    v16bf af[2], bf[4];
#pragma unroll
    for (int mf = 0; mf < 2; ++mf) af[mf] = load_fragA(&lA[buf][0][0], mIdx * 32 + mf * 16, lane);
    const int bCol[4] = { nIdx * 32, nIdx * 32 + 16, 64 + nIdx * 32, 64 + nIdx * 32 + 16 };
#pragma unroll
    for (int nf = 0; nf < 4; ++nf) bf[nf] = load_fragB(&lB[buf][0][0], bCol[nf], lane);
#pragma unroll
    for (int mf = 0; mf < 2; ++mf)
#pragma unroll
      for (int nf = 0; nf < 4; ++nf) acc[mf][nf] = wmma_bf16(af[mf], bf[nf], acc[mf][nf]);
    buf ^= 1;
  }
  // fused gate epilogue: frags j (z) pair with j+2 (h~) at the same columns
#pragma unroll
  for (int mf = 0; mf < 2; ++mf)
#pragma unroll
    for (int j = 0; j < 2; ++j) {
      const int col = n0 + nIdx * 32 + j * 16 + (lane & 15);
      const float bz = bias[col], bh = bias[DD + col];
#pragma unroll
      for (int rr = 0; rr < 8; ++rr) {
        const size_t row = m0 + mIdx * 32 + mf * 16 + ((lane < 16) ? rr : rr + 8);
        const float z  = sigm(acc[mf][j][rr] + bz);
        const float ht = acc[mf][j + 2][rr] + bh;
        Aout[row * DD + col] = 1.0f - z;
        Bout[row * DD + col] = z * ht;
      }
    }
}

// ============================================================
// Chunked associative scan: h_t = a_t * h_{t-1} + bb_t  along T.
// ============================================================
__global__ void scan_phaseA(const float* __restrict__ A, const float* __restrict__ Bv,
                            float* __restrict__ Aprod, float* __restrict__ Hend) {
  const int d = (blockIdx.x & 3) * 256 + threadIdx.x;  // D/256 = 4
  const int c = (blockIdx.x >> 2) & (NCHK - 1);
  const int b = blockIdx.x >> 8;
  size_t base = ((size_t)(b * TT + c * CL)) * DD + d;
  float ap = 1.0f, h = 0.0f;
  for (int t = 0; t < CL; ++t) {
    const float av = A[base + (size_t)t * DD], bv = Bv[base + (size_t)t * DD];
    ap *= av;
    h = av * h + bv;
  }
  const size_t ci = ((size_t)b * NCHK + c) * DD + d;
  Aprod[ci] = ap;
  Hend[ci]  = h;
}

__global__ void scan_carry(const float* __restrict__ Aprod, const float* __restrict__ Hend,
                           float* __restrict__ CarryIn) {
  const int d = (blockIdx.x & 3) * 256 + threadIdx.x;
  const int b = blockIdx.x >> 2;
  float carry = 0.0f;
  for (int c = 0; c < NCHK; ++c) {
    const size_t ci = ((size_t)b * NCHK + c) * DD + d;
    CarryIn[ci] = carry;                       // h entering chunk c
    carry = Aprod[ci] * carry + Hend[ci];
  }
}

__global__ void scan_phaseC(const float* __restrict__ A, const float* __restrict__ Bv,
                            const float* __restrict__ CarryIn, float* __restrict__ H) {
  const int d = (blockIdx.x & 3) * 256 + threadIdx.x;
  const int c = (blockIdx.x >> 2) & (NCHK - 1);
  const int b = blockIdx.x >> 8;
  size_t base = ((size_t)(b * TT + c * CL)) * DD + d;
  float h = CarryIn[((size_t)b * NCHK + c) * DD + d];
  for (int t = 0; t < CL; ++t) {
    const float av = A[base + (size_t)t * DD], bv = Bv[base + (size_t)t * DD];
    h = av * h + bv;
    H[base + (size_t)t * DD] = h;
  }
}

// ============================================================
// Post-GEMM: out[b][c][t] = (h @ w_post)[b][t][c] + b_post[c]
// M tile 128, N tile 16; 8 waves of 16 M-rows.
// ============================================================
__global__ __launch_bounds__(256)
void gemm_post(const float* __restrict__ H, const __bf16* __restrict__ Wt,
               const float* __restrict__ bias, float* __restrict__ Out) {
  const int tid = threadIdx.x, lane = tid & 31, wv = tid >> 5;
  const size_t m0 = (size_t)blockIdx.x * 128;
  const int n0 = blockIdx.y * 16;

  __shared__ __align__(16) __bf16 lA[128][LDA];
  __shared__ __align__(16) __bf16 lBp[16][LDA];

  v8f acc = {};
  for (int kt = 0; kt < DD / 32; ++kt) {
    const int k0 = kt * 32;
    __syncthreads();
    { // A tile
      const int r = tid >> 1, half = tid & 1;
      const float* src = H + (m0 + r) * DD + k0 + half * 16;
      float v[16];
      float4 f0 = *(const float4*)(src);
      float4 f1 = *(const float4*)(src + 4);
      float4 f2 = *(const float4*)(src + 8);
      float4 f3 = *(const float4*)(src + 12);
      v[0] = f0.x;  v[1] = f0.y;  v[2]  = f0.z;  v[3]  = f0.w;
      v[4] = f1.x;  v[5] = f1.y;  v[6]  = f1.z;  v[7]  = f1.w;
      v[8] = f2.x;  v[9] = f2.y;  v[10] = f2.z;  v[11] = f2.w;
      v[12] = f3.x; v[13] = f3.y; v[14] = f3.z;  v[15] = f3.w;
      v8bf p0, p1; pack16(v, p0, p1);
      *(v8bf*)&lA[r][half * 16]     = p0;
      *(v8bf*)&lA[r][half * 16 + 8] = p1;
    }
    if (tid < 32) { // B tile: 16 rows of Wt[NPOST][DD]
      const int rr = tid >> 1, half = tid & 1;
      const __bf16* src = Wt + (size_t)(n0 + rr) * DD + k0 + half * 16;
      uint4 u0 = *(const uint4*)(src);
      uint4 u1 = *(const uint4*)(src + 8);
      *(uint4*)&lBp[rr][half * 16]     = u0;
      *(uint4*)&lBp[rr][half * 16 + 8] = u1;
    }
    __syncthreads();
    v16bf af = load_fragA(&lA[0][0], wv * 16, lane);
    v16bf bf = load_fragB(&lBp[0][0], 0, lane);
    acc = wmma_bf16(af, bf, acc);
  }
  const int col = n0 + (lane & 15);
  if (col < COUT) {
    const float bs = bias[col];
#pragma unroll
    for (int rr = 0; rr < 8; ++rr) {
      const size_t row = m0 + wv * 16 + ((lane < 16) ? rr : rr + 8);
      const size_t b = row / TT, t = row % TT;
      Out[(b * COUT + col) * TT + t] = acc[rr] + bs;
    }
  }
}

// ============================================================
// Host launcher
// ============================================================
extern "C" void kernel_launch(void* const* d_in, const int* in_sizes, int n_in,
                              void* d_out, int out_size, void* d_ws, size_t ws_size,
                              hipStream_t stream) {
  (void)in_sizes; (void)n_in; (void)out_size; (void)ws_size;
  const float* x        = (const float*)d_in[0];
  const float* w_pre    = (const float*)d_in[1];
  const float* b_pre    = (const float*)d_in[2];
  const float* w_layers = (const float*)d_in[3];
  const float* b_layers = (const float*)d_in[4];
  const float* w_post   = (const float*)d_in[5];
  const float* b_post   = (const float*)d_in[6];
  float* out = (float*)d_out;

  // workspace layout (floats, then bf16)
  const size_t HB = (size_t)BB * TT * DD;          // 67,108,864
  const size_t CH = (size_t)BB * NCHK * DD;        // 1,048,576
  float* h     = (float*)d_ws;
  float* aBuf  = h + HB;
  float* bBuf  = aBuf + HB;
  float* Aprod = bBuf + HB;
  float* Hend  = Aprod + CH;
  float* Carry = Hend + CH;
  __bf16* wtL    = (__bf16*)(Carry + CH);
  __bf16* wtPre  = wtL + (size_t)LL * 2 * DD * DD;
  __bf16* wtPost = wtPre + (size_t)DD * KPRE;

  // 1) weight prep (bf16, transposed)
  prep_layer_w<<<(LL * 2 * DD * DD) / 256, 256, 0, stream>>>(w_layers, wtL);
  prep_pre_w<<<(DD * KPRE) / 256, 256, 0, stream>>>(w_pre, wtPre);
  prep_post_w<<<(NPOST * DD) / 256, 256, 0, stream>>>(w_post, wtPost);

  const int MT = (BB * TT) / 128;  // 512 M-tiles

  // 2) h = x @ w_pre + b_pre
  gemm_pre<<<dim3(MT, DD / 128), 256, 0, stream>>>(x, wtPre, b_pre, h);

  // 3) minGRU layers
  for (int l = 0; l < LL; ++l) {
    gemm_layer<<<dim3(MT, DD / 64), 256, 0, stream>>>(
        h, wtL + (size_t)l * 2 * DD * DD, b_layers + (size_t)l * 2 * DD, aBuf, bBuf);
    scan_phaseA<<<BB * NCHK * (DD / 256), 256, 0, stream>>>(aBuf, bBuf, Aprod, Hend);
    scan_carry<<<BB * (DD / 256), 256, 0, stream>>>(Aprod, Hend, Carry);
    scan_phaseC<<<BB * NCHK * (DD / 256), 256, 0, stream>>>(aBuf, bBuf, Carry, h);
  }

  // 4) out = (h @ w_post + b_post), transposed to [b][c][t]
  gemm_post<<<dim3(MT, NPOST / 16), 256, 0, stream>>>(h, wtPost, b_post, out);
}
